// MPMC_net_90804198572793
// MI455X (gfx1250) — matrix operand order repaired and checked by reading, not compiled, not run on hardware
//
#include <hip/hip_runtime.h>

#define NTOT    16384
#define NHID    128
#define NBATCH  16
#define NSAMP   1024
#define NLAYERS 3

typedef _Float16 v16h __attribute__((ext_vector_type(16)));
typedef _Float16 v8h  __attribute__((ext_vector_type(8)));
typedef float    v8f  __attribute__((ext_vector_type(8)));

// LDS row strides (f16 elements), padded +8 to rotate banks per row
#define LDA1 264   // 256-wide A panel
#define LDA2 136   // 128-wide M panel

// ---------------------------------------------------------------------------
// WMMA K-loop: Ash is a row-major f16 LDS panel (16 rows x (KT*32) cols),
// Wp is pre-packed B in fragment order: ((kt*8 + nt)*32 + lane)*16 f16 each.
// A fragment per ISA 7.12.2 (16-bit A 16x32): lane<16 -> K 0..7 / 16..23,
// lane>=16 -> K 8..15 / 24..31 of row M=lane%16.
// ---------------------------------------------------------------------------
template<int KT>
__device__ __forceinline__ v8f wmma_kloop(const _Float16* Ash, int lda,
                                          const _Float16* Wp, int nt, int lane,
                                          v8f acc)
{
  const int arow = lane & 15;
  const int koff = (lane >> 4) * 8;
#pragma unroll
  for (int kt = 0; kt < KT; ++kt) {
    const int kb = kt * 32 + koff;
    v8h lo = *(const v8h*)(Ash + arow * lda + kb);
    v8h hi = *(const v8h*)(Ash + arow * lda + kb + 16);
    v16h a = __builtin_shufflevector(lo, hi,
                                     0,1,2,3,4,5,6,7,8,9,10,11,12,13,14,15);
    const _Float16* bp = Wp + ((size_t)((kt * 8 + nt) * 32 + lane)) * 16;
    v16h b = *(const v16h*)bp;
    if (kt + 1 < KT)
      __builtin_prefetch(Wp + ((size_t)(((kt + 1) * 8 + nt) * 32 + lane)) * 16, 0, 0);
    acc = __builtin_amdgcn_wmma_f32_16x16x32_f16(
        /*neg_a=*/false, a, /*neg_b=*/false, b,
        /*c_mod=*/(short)0, acc, /*reuse_a=*/false, /*reuse_b=*/false);
  }
  return acc;
}

// Two-layer MLP on a 16-row LDS panel: relu(relu(A@W1+b1)@W2+b2).
// Block = 128 threads (4 waves). Wave w owns N-tiles {w, w+4}.
// KT1 k-tiles for GEMM1 (A panel width KT1*32), GEMM2 always K=128 (4 tiles).
template<int KT1>
__device__ __forceinline__ void mlp2_tile(const _Float16* Ash, _Float16* Msh,
                                          const _Float16* Wp1, const float* b1,
                                          const _Float16* Wp2, const float* b2,
                                          int lane, int wv, v8f& o0, v8f& o1)
{
  const int ncol0 = wv * 16 + (lane & 15);
  const int ncol1 = ncol0 + 64;
  const int mbase = (lane >> 4) * 8;

  v8f acc0, acc1;
  {
    float bv0 = b1[ncol0], bv1 = b1[ncol1];
#pragma unroll
    for (int r = 0; r < 8; ++r) { acc0[r] = bv0; acc1[r] = bv1; }
  }
  acc0 = wmma_kloop<KT1>(Ash, LDA1, Wp1, wv,     lane, acc0);
  acc1 = wmma_kloop<KT1>(Ash, LDA1, Wp1, wv + 4, lane, acc1);

  // ReLU, convert to f16, deposit into M panel (row-major) for GEMM2's A.
#pragma unroll
  for (int r = 0; r < 8; ++r) {
    float v0 = fmaxf(acc0[r], 0.0f);
    float v1 = fmaxf(acc1[r], 0.0f);
    Msh[(mbase + r) * LDA2 + ncol0] = (_Float16)v0;
    Msh[(mbase + r) * LDA2 + ncol1] = (_Float16)v1;
  }
  __syncthreads();

  {
    float bv0 = b2[ncol0], bv1 = b2[ncol1];
#pragma unroll
    for (int r = 0; r < 8; ++r) { acc0[r] = bv0; acc1[r] = bv1; }
  }
  acc0 = wmma_kloop<4>(Msh, LDA2, Wp2, wv,     lane, acc0);
  acc1 = wmma_kloop<4>(Msh, LDA2, Wp2, wv + 4, lane, acc1);
#pragma unroll
  for (int r = 0; r < 8; ++r) {
    o0[r] = fmaxf(acc0[r], 0.0f);
    o1[r] = fmaxf(acc1[r], 0.0f);
  }
}

// ---------------------------------------------------------------------------
// Pack f32 weights [K,128] row-major -> f16 B fragments.
// packed[((kt*8+nt)*32+lane)*16 + e] = W[kt*32 + (lane>>4)*8 + (e&7) + (e>>3)*16][nt*16 + lane%16]
// Layout in Wp: msg1@0 (3x32768), msg2@98304 (3x16384),
//               upd1@147456 (3x32768), upd2@245760 (3x16384). Total 294912.
// ---------------------------------------------------------------------------
__global__ void pack_weights_kernel(const float* __restrict__ msg1_W,
                                    const float* __restrict__ msg2_W,
                                    const float* __restrict__ upd1_W,
                                    const float* __restrict__ upd2_W,
                                    _Float16* __restrict__ Wp)
{
  int p = blockIdx.x * blockDim.x + threadIdx.x;
  if (p >= 294912) return;
  const float* W;
  int K, q = p;
  if      (q <  98304) { W = msg1_W; K = 256; }
  else if (q < 147456) { q -=  98304; W = msg2_W; K = 128; }
  else if (q < 245760) { q -= 147456; W = upd1_W; K = 256; }
  else                 { q -= 245760; W = upd2_W; K = 128; }
  const int perLayer = K * NHID;
  const int l = q / perLayer;
  const int r = q % perLayer;
  const int e    =  r        & 15;
  const int lane = (r >>  4) & 31;
  const int nt   = (r >>  9) &  7;
  const int kt   =  r >> 12;
  const int k = kt * 32 + ((lane >> 4) << 3) + (e & 7) + ((e >> 3) << 4);
  const int n = nt * 16 + (lane & 15);
  Wp[p] = (_Float16)W[(size_t)l * perLayer + (size_t)k * NHID + n];
}

// h = X @ enc_W + enc_b  (N x 3 @ 3 x 128), stored f16
__global__ void encoder_kernel(const float* __restrict__ X,
                               const float* __restrict__ W,
                               const float* __restrict__ b,
                               _Float16* __restrict__ h16)
{
  const int n = blockIdx.x;
  const int c = threadIdx.x;
  const float x0 = X[n * 3 + 0], x1 = X[n * 3 + 1], x2 = X[n * 3 + 2];
  float v = b[c] + x0 * W[c] + x1 * W[NHID + c] + x2 * W[2 * NHID + c];
  h16[(size_t)n * NHID + c] = (_Float16)v;
}

// Edge MLP: m = relu(relu([h[dst],h[src]]@W1+b1)@W2+b2); agg[dst] += m
// Block = 16 edges, 128 threads.
__global__ void edge_mlp_kernel(const _Float16* __restrict__ h16,
                                const int* __restrict__ ei, int E,
                                const _Float16* __restrict__ Wp1,
                                const float* __restrict__ b1,
                                const _Float16* __restrict__ Wp2,
                                const float* __restrict__ b2,
                                float* __restrict__ agg)
{
  __shared__ __align__(16) _Float16 Ash[16 * LDA1];
  __shared__ __align__(16) _Float16 Msh[16 * LDA2];
  __shared__ int dst_sh[16];

  const int tid  = threadIdx.x;
  const int lane = tid & 31;
  const int wv   = tid >> 5;
  const int ebase = blockIdx.x * 16;
  const int* __restrict__ src = ei;
  const int* __restrict__ dst = ei + E;

  // Stage A panel: cols [0,128) = h[dst[e]], cols [128,256) = h[src[e]]
  {
    const int row = tid >> 3;          // 0..15 (edge within tile)
    const int seg = tid & 7;           // 8 threads x 32 f16 per row
    int eg = ebase + row;
    if (eg > E - 1) eg = E - 1;
    const int node = (seg < 4) ? dst[eg] : src[eg];
    if (seg == 0) dst_sh[row] = dst[eg];
    const int csrc = (seg & 3) * 32;
    const _Float16* hr = h16 + (size_t)node * NHID + csrc;
    _Float16* ar = Ash + row * LDA1 + ((seg < 4) ? 0 : 128) + csrc;
#pragma unroll
    for (int j = 0; j < 4; ++j)
      *(v8h*)(ar + j * 8) = *(const v8h*)(hr + j * 8);
  }
  __syncthreads();

  v8f o0, o1;
  mlp2_tile<8>(Ash, Msh, Wp1, b1, Wp2, b2, lane, wv, o0, o1);

  // Scatter-add into agg[dst] (segment_sum)
  const int ncol0 = wv * 16 + (lane & 15);
#pragma unroll
  for (int r = 0; r < 8; ++r) {
    const int m = r + ((lane >> 4) << 3);
    if (ebase + m < E) {
      const int node = dst_sh[m];
      atomicAdd(agg + (size_t)node * NHID + ncol0,      o0[r]);
      atomicAdd(agg + (size_t)node * NHID + ncol0 + 64, o1[r]);
    }
  }
}

// Node update: hpre = relu(relu([h,agg]@W1+b1)@W2+b2). Block = 16 nodes.
// hpre may alias agg: each block reads its rows into LDS before writing them.
__global__ void update_mlp_kernel(const _Float16* __restrict__ h16,
                                  const float* __restrict__ agg,
                                  const _Float16* __restrict__ Wp1,
                                  const float* __restrict__ b1,
                                  const _Float16* __restrict__ Wp2,
                                  const float* __restrict__ b2,
                                  float* __restrict__ hpre)
{
  __shared__ __align__(16) _Float16 Ash[16 * LDA1];
  __shared__ __align__(16) _Float16 Msh[16 * LDA2];

  const int tid  = threadIdx.x;
  const int lane = tid & 31;
  const int wv   = tid >> 5;
  const int nbase = blockIdx.x * 16;

  // Stage A panel: cols [0,128) = h16[node], cols [128,256) = f16(agg[node])
  {
    const int row = tid >> 3;
    const int seg = tid & 7;
    const int node = nbase + row;
    const int csrc = (seg & 3) * 32;
    if (seg < 4) {
      const _Float16* hr = h16 + (size_t)node * NHID + csrc;
      _Float16* ar = Ash + row * LDA1 + csrc;
#pragma unroll
      for (int j = 0; j < 4; ++j)
        *(v8h*)(ar + j * 8) = *(const v8h*)(hr + j * 8);
    } else {
      const float* gr = agg + (size_t)node * NHID + csrc;
      _Float16* ar = Ash + row * LDA1 + 128 + csrc;
#pragma unroll
      for (int j = 0; j < 32; ++j) ar[j] = (_Float16)gr[j];
    }
  }
  __syncthreads();

  v8f o0, o1;
  mlp2_tile<8>(Ash, Msh, Wp1, b1, Wp2, b2, lane, wv, o0, o1);

  const int ncol0 = wv * 16 + (lane & 15);
#pragma unroll
  for (int r = 0; r < 8; ++r) {
    const int node = nbase + r + ((lane >> 4) << 3);
    hpre[(size_t)node * NHID + ncol0]      = o0[r];
    hpre[(size_t)node * NHID + ncol0 + 64] = o1[r];
  }
}

// Per (graph, channel) normalization over NSAMP samples (ddof=0), write f16.
__global__ void layernorm_kernel(const float* __restrict__ hpre,
                                 _Float16* __restrict__ h16)
{
  const int g = blockIdx.x >> 7;        // / NHID
  const int c = blockIdx.x & (NHID - 1);
  const int t = threadIdx.x;            // 128 threads
  __shared__ float s_sum[128], s_sq[128];
  const float* p = hpre + (size_t)g * NSAMP * NHID + c;
  float s = 0.f, q = 0.f;
  for (int i = t; i < NSAMP; i += 128) {
    float v = p[(size_t)i * NHID];
    s += v; q += v * v;
  }
  s_sum[t] = s; s_sq[t] = q;
  __syncthreads();
  for (int off = 64; off > 0; off >>= 1) {
    if (t < off) { s_sum[t] += s_sum[t + off]; s_sq[t] += s_sq[t + off]; }
    __syncthreads();
  }
  const float inv_n = 1.0f / (float)NSAMP;
  const float mu  = s_sum[0] * inv_n;
  const float var = s_sq[0] * inv_n - mu * mu;
  const float rs  = rsqrtf(var + 1e-5f);
  _Float16* o = h16 + (size_t)g * NSAMP * NHID + c;
  for (int i = t; i < NSAMP; i += 128) {
    float v = p[(size_t)i * NHID];
    o[(size_t)i * NHID] = (_Float16)((v - mu) * rs);
  }
}

// Xr = sigmoid(h @ dec_W + dec_b)  (N x 128 @ 128 x 3)
__global__ void decoder_kernel(const _Float16* __restrict__ h16,
                               const float* __restrict__ W,
                               const float* __restrict__ b,
                               float* __restrict__ Xr)
{
  const int n = blockIdx.x * blockDim.x + threadIdx.x;
  if (n >= NTOT) return;
  float a0 = b[0], a1 = b[1], a2 = b[2];
  const _Float16* hr = h16 + (size_t)n * NHID;
#pragma unroll 4
  for (int k = 0; k < NHID; ++k) {
    const float hv = (float)hr[k];
    a0 += hv * W[k * 3 + 0];
    a1 += hv * W[k * 3 + 1];
    a2 += hv * W[k * 3 + 2];
  }
  Xr[n * 3 + 0] = 1.0f / (1.0f + __expf(-a0));
  Xr[n * 3 + 1] = 1.0f / (1.0f + __expf(-a1));
  Xr[n * 3 + 2] = 1.0f / (1.0f + __expf(-a2));
}

// Pairwise discrepancy terms: block per (g, i); sums[g]=sum1, sums[16+g]=sum2.
__global__ void disc_pairs_kernel(const float* __restrict__ Xr,
                                  float* __restrict__ sums)
{
  const int g = blockIdx.x >> 10;
  const int i = blockIdx.x & (NSAMP - 1);
  const float* base = Xr + (size_t)g * NSAMP * 3;
  const float xi0 = base[i * 3 + 0];
  const float xi1 = base[i * 3 + 1];
  const float xi2 = base[i * 3 + 2];
  float acc = 0.f;
  for (int j = threadIdx.x; j < NSAMP; j += 256) {
    const float a = fmaxf(xi0, base[j * 3 + 0]);
    const float bb = fmaxf(xi1, base[j * 3 + 1]);
    const float c = fmaxf(xi2, base[j * 3 + 2]);
    acc += (1.f - a) * (1.f - bb) * (1.f - c);
  }
  __shared__ float sh[256];
  sh[threadIdx.x] = acc;
  __syncthreads();
  for (int off = 128; off > 0; off >>= 1) {
    if (threadIdx.x < off) sh[threadIdx.x] += sh[threadIdx.x + off];
    __syncthreads();
  }
  if (threadIdx.x == 0) {
    atomicAdd(sums + NBATCH + g, sh[0]);
    const float p1 = (1.f - xi0 * xi0) * (1.f - xi1 * xi1) * (1.f - xi2 * xi2);
    atomicAdd(sums + g, p1);
  }
}

__global__ void loss_kernel(const float* __restrict__ sums,
                            float* __restrict__ out)
{
  __shared__ float sh[NBATCH];
  const int t = threadIdx.x;
  if (t < NBATCH) {
    const float inv_n = 1.0f / (float)NSAMP;
    float d = (1.0f / 27.0f) - inv_n * 0.25f * sums[t]
            + inv_n * inv_n * sums[NBATCH + t];
    sh[t] = sqrtf(fmaxf(d, 0.0f));
  }
  __syncthreads();
  if (t == 0) {
    float s = 0.f;
    for (int g = 0; g < NBATCH; ++g) s += sh[g];
    out[0] = s / (float)NBATCH;
  }
}

// ---------------------------------------------------------------------------
extern "C" void kernel_launch(void* const* d_in, const int* in_sizes, int n_in,
                              void* d_out, int out_size, void* d_ws, size_t ws_size,
                              hipStream_t stream)
{
  (void)n_in; (void)out_size; (void)ws_size;
  const float* X      = (const float*)d_in[0];
  const int*   ei     = (const int*)  d_in[2];
  const float* enc_W  = (const float*)d_in[3];
  const float* enc_b  = (const float*)d_in[4];
  const float* msg1_W = (const float*)d_in[5];
  const float* msg1_b = (const float*)d_in[6];
  const float* msg2_W = (const float*)d_in[7];
  const float* msg2_b = (const float*)d_in[8];
  const float* upd1_W = (const float*)d_in[9];
  const float* upd1_b = (const float*)d_in[10];
  const float* upd2_W = (const float*)d_in[11];
  const float* upd2_b = (const float*)d_in[12];
  const float* dec_W  = (const float*)d_in[13];
  const float* dec_b  = (const float*)d_in[14];
  const int E = in_sizes[2] / 2;

  // Workspace layout (base is 256-aligned from hipMalloc)
  char* ws = (char*)d_ws;
  _Float16* h16 = (_Float16*)ws;                               // 4,194,304 B
  float*    agg = (float*)(ws + 4194304);                      // 8,388,608 B
  _Float16* Wp  = (_Float16*)(ws + 4194304 + 8388608);         //   589,824 B
  float*    sums = (float*)(ws + 4194304 + 8388608 + 589824);  //       128 B

  float* out = (float*)d_out;
  float* Xr  = out + 1;   // (loss, Xr): loss at [0], Xr flat after

  pack_weights_kernel<<<1152, 256, 0, stream>>>(msg1_W, msg2_W, upd1_W, upd2_W, Wp);
  encoder_kernel<<<NTOT, NHID, 0, stream>>>(X, enc_W, enc_b, h16);

  const int eblocks = (E + 15) / 16;
  for (int l = 0; l < NLAYERS; ++l) {
    hipMemsetAsync(agg, 0, (size_t)NTOT * NHID * sizeof(float), stream);
    edge_mlp_kernel<<<eblocks, 128, 0, stream>>>(
        h16, ei, E,
        Wp + (size_t)l * 32768,          msg1_b + l * NHID,
        Wp + 98304 + (size_t)l * 16384,  msg2_b + l * NHID,
        agg);
    update_mlp_kernel<<<NTOT / 16, 128, 0, stream>>>(
        h16, agg,
        Wp + 147456 + (size_t)l * 32768, upd1_b + l * NHID,
        Wp + 245760 + (size_t)l * 16384, upd2_b + l * NHID,
        agg /* hpre aliases agg: safe, rows staged before overwrite */);
    layernorm_kernel<<<NBATCH * NHID, 128, 0, stream>>>(agg, h16);
  }

  decoder_kernel<<<NTOT / 256, 256, 0, stream>>>(h16, dec_W, dec_b, Xr);
  hipMemsetAsync(sums, 0, 2 * NBATCH * sizeof(float), stream);
  disc_pairs_kernel<<<NBATCH * NSAMP, 256, 0, stream>>>(Xr, sums);
  loss_kernel<<<1, 32, 0, stream>>>(sums, out);
}
